// Model2_8340826488964
// MI455X (gfx1250) — matrix operand order repaired and checked by reading, not compiled, not run on hardware
//
#include <hip/hip_runtime.h>
#include <hip/hip_bf16.h>

#define NH 1024
#define NB 2048
#define NT 512
#define NV 31
#define BN_EPS 1e-5f

typedef __attribute__((ext_vector_type(16))) __bf16 bf16x16;
typedef __attribute__((ext_vector_type(8)))  __bf16 bf16x8;
typedef __attribute__((ext_vector_type(8)))  float  v8f;

__device__ __forceinline__ bf16x16 concat8(bf16x8 lo, bf16x8 hi) {
    return __builtin_shufflevector(lo, hi, 0,1,2,3,4,5,6,7,8,9,10,11,12,13,14,15);
}

// A-matrix bf16 fragment (16x32): lane half selects K {0-7,16-23} vs {8-15,24-31}
__device__ __forceinline__ bf16x16 frag_A(const __bf16* row, int khalf) {
    bf16x8 lo = *(const bf16x8*)(row + khalf * 8);
    bf16x8 hi = *(const bf16x8*)(row + 16 + khalf * 8);
    return concat8(lo, hi);
}

// B-matrix bf16 fragment (32x16), row ptr = column n of B stored K-contiguous:
// lane half selects K {0-7,16-23} vs {8-15,24-31}
__device__ __forceinline__ bf16x16 frag_B(const __bf16* row, int khalf) {
    const __bf16* p = row + khalf * 8;
    bf16x8 lo = *(const bf16x8*)p;
    bf16x8 hi = *(const bf16x8*)(p + 16);
    return concat8(lo, hi);
}

__device__ __forceinline__ v8f wmma_bf16(bf16x16 a, bf16x16 b, v8f c) {
    return __builtin_amdgcn_wmma_f32_16x16x32_bf16(false, a, false, b, (short)0, c,
                                                   false, false);
}

// CDNA5 async memory->LDS copy (16B per lane), tracked by ASYNCcnt.
// GVS addressing: lds_byte_addr(VGPR), byte_offset(VGPR), base(SGPR pair).
__device__ __forceinline__ void async_ld_b128(void* lds_ptr, unsigned byte_off,
                                              const void* base) {
    unsigned lds_addr = (unsigned)(size_t)lds_ptr;  // low 32 bits = LDS offset
    asm volatile("global_load_async_to_lds_b128 %0, %1, %2"
                 :
                 : "v"(lds_addr), "v"(byte_off), "s"(base)
                 : "memory");
}

__device__ __forceinline__ void wait_asynccnt0() {
    asm volatile("s_wait_asynccnt 0x0" ::: "memory");
}

// ---------------------------------------------------------------- init kernel
__global__ __launch_bounds__(256) void init_ws(const float* __restrict__ W_hh,
                                               const float* __restrict__ W_ho,
                                               __bf16* __restrict__ Whh_b,
                                               __bf16* __restrict__ Who_b,
                                               float* __restrict__ hA) {
    size_t i = (size_t)blockIdx.x * blockDim.x + threadIdx.x;  // covers 2M
    if (i < (size_t)NH * NH) Whh_b[i] = (__bf16)W_hh[i];
    if (i < (size_t)32 * NH) {
        int j = (int)(i >> 10), k = (int)(i & 1023);
        Who_b[i] = (j < NV) ? (__bf16)W_ho[j * NH + k] : (__bf16)0.0f;
    }
    if (i < (size_t)NB * NH) hA[i] = 0.0f;
}

// ------------------------------------------------------- main recurrent GEMM
// h_out = relu( (h_in + emb[x[:,t]]) @ W_hh.T + b_hh )
// block tile 128x128, 8 waves, wave tile 32x64 (2x4 wmma accumulators).
// Double-buffered LDS; B tile staged with GLOBAL_LOAD_ASYNC_TO_LDS_B128,
// A tile fused (emb gather + add + bf16 cvt) on the VALU path.
__global__ __launch_bounds__(256) void gemm_hh(const float* __restrict__ h_in,
                                               float* __restrict__ h_out,
                                               const int* __restrict__ x, int t,
                                               const float* __restrict__ emb,
                                               const __bf16* __restrict__ Wb,
                                               const float* __restrict__ bhh) {
    __shared__ alignas(16) __bf16 As[2][128][32];  // A tiles (M x K), bf16
    __shared__ alignas(16) __bf16 Bs[2][128][32];  // Bs[n][k] = W[n0+n][k0+k]
    __shared__ int xid[128];

    const int m0 = blockIdx.x * 128;
    const int n0 = blockIdx.y * 128;
    const int tid = threadIdx.x;
    const int wave = tid >> 5;
    const int lane = tid & 31;
    const int lane15 = lane & 15;
    const int khalf = lane >> 4;
    const int wm = (wave & 3) * 32;  // wave M offset in block tile
    const int wn = (wave >> 2) * 64; // wave N offset in block tile

    if (tid < 128) xid[tid] = x[(size_t)(m0 + tid) * NT + t];
    __syncthreads();

    auto stage = [&](int buf, int k0) {
        // --- B tile: pure copy, async memory -> LDS (2 x b128 per thread) ---
#pragma unroll
        for (int i = 0; i < 2; ++i) {
            int slot = tid + i * 256;  // 0..511 slots of 16 bytes
            int row = slot >> 2;
            int seg = (slot & 3) * 8;  // bf16 units
            unsigned goff = (unsigned)(((n0 + row) * NH + k0 + seg) * 2);
            async_ld_b128(&Bs[buf][row][seg], goff, Wb);
        }
        // --- A tile: fused embedding add + bf16 convert (128x32) ---
#pragma unroll
        for (int i = 0; i < 2; ++i) {
            int slot = tid + i * 256;
            int row = slot >> 2;
            int seg = (slot & 3) * 8;
            const float* hp = h_in + (size_t)(m0 + row) * NH + k0 + seg;
            const float* ep = emb + (size_t)xid[row] * NH + k0 + seg;
            float4 ha = *(const float4*)hp;
            float4 hb = *(const float4*)(hp + 4);
            float4 ea = *(const float4*)ep;
            float4 eb = *(const float4*)(ep + 4);
            union { __bf16 e[8]; bf16x8 v; } u;
            u.e[0] = (__bf16)(ha.x + ea.x); u.e[1] = (__bf16)(ha.y + ea.y);
            u.e[2] = (__bf16)(ha.z + ea.z); u.e[3] = (__bf16)(ha.w + ea.w);
            u.e[4] = (__bf16)(hb.x + eb.x); u.e[5] = (__bf16)(hb.y + eb.y);
            u.e[6] = (__bf16)(hb.z + eb.z); u.e[7] = (__bf16)(hb.w + eb.w);
            *(bf16x8*)&As[buf][row][seg] = u.v;
        }
    };

    v8f acc[2][4] = {};

    stage(0, 0);
    wait_asynccnt0();
    __syncthreads();

    for (int k0 = 0; k0 < NH; k0 += 32) {
        const int cur = (k0 >> 5) & 1;
        if (k0 + 32 < NH) stage(cur ^ 1, k0 + 32);  // overlap with WMMA below

        bf16x16 af[2], bf_[4];
#pragma unroll
        for (int i = 0; i < 2; ++i)
            af[i] = frag_A(&As[cur][wm + i * 16 + lane15][0], khalf);
#pragma unroll
        for (int j = 0; j < 4; ++j)
            bf_[j] = frag_B(&Bs[cur][wn + j * 16 + lane15][0], khalf);
#pragma unroll
        for (int i = 0; i < 2; ++i)
#pragma unroll
            for (int j = 0; j < 4; ++j) acc[i][j] = wmma_bf16(af[i], bf_[j], acc[i][j]);

        wait_asynccnt0();   // async B copy for next tile complete
        __syncthreads();    // + all waves done reading current tiles
    }

    // epilogue: bias + relu -> h_out
#pragma unroll
    for (int i = 0; i < 2; ++i) {
#pragma unroll
        for (int j = 0; j < 4; ++j) {
            int n = n0 + wn + j * 16 + lane15;
            int mb = m0 + wm + i * 16 + khalf * 8;
            float bias = bhh[n];
            v8f c = acc[i][j];
#pragma unroll
            for (int r = 0; r < 8; ++r) {
                float v = c[r] + bias;
                h_out[(size_t)(mb + r) * NH + n] = v > 0.0f ? v : 0.0f;
            }
        }
    }
}

// --------------------------------------------- batch-norm stats (per feature)
__global__ __launch_bounds__(256) void bn_stats(const float* __restrict__ h,
                                                const float* __restrict__ gamma,
                                                const float* __restrict__ beta,
                                                float* __restrict__ scale,
                                                float* __restrict__ shift) {
    __shared__ float ssum[8][32];
    __shared__ float ssq[8][32];
    const int f = threadIdx.x & 31;
    const int r = threadIdx.x >> 5;
    const int feat = blockIdx.x * 32 + f;
    float sum = 0.0f, sq = 0.0f;
    for (int b = r; b < NB; b += 8) {
        float v = h[(size_t)b * NH + feat];
        sum += v;
        sq += v * v;
    }
    ssum[r][f] = sum;
    ssq[r][f] = sq;
    __syncthreads();
    if (r == 0) {
#pragma unroll
        for (int i = 1; i < 8; ++i) { sum += ssum[i][f]; sq += ssq[i][f]; }
        float mu = sum * (1.0f / NB);
        float var = sq * (1.0f / NB) - mu * mu;
        float sc = gamma[feat] * rsqrtf(var + BN_EPS);
        scale[feat] = sc;
        shift[feat] = beta[feat] - mu * sc;
    }
}

// ------------------------------------------ normalized output GEMM (N=31->32)
// out[:, t, :] = ((h*scale+shift) @ W_ho.T + b_ho)
__global__ __launch_bounds__(256) void out_gemm(const float* __restrict__ h,
                                                const float* __restrict__ scale,
                                                const float* __restrict__ shift,
                                                const __bf16* __restrict__ Wob,
                                                const float* __restrict__ bho,
                                                float* __restrict__ out, int t) {
    const int tid = threadIdx.x;
    const int wave = tid >> 5;
    const int lane = tid & 31;
    const int lane15 = lane & 15;
    const int khalf = lane >> 4;
    const int mrow0 = blockIdx.x * 128 + wave * 16;  // 16 rows per wave
    const float* hrow = h + (size_t)(mrow0 + lane15) * NH;

    v8f acc[2] = {};

    for (int k0 = 0; k0 < NH; k0 += 32) {
        // A fragment: bf16(h*scale + shift), built in registers
        union { __bf16 e[16]; bf16x16 v; } af;
#pragma unroll
        for (int half = 0; half < 2; ++half) {
            int kk = k0 + half * 16 + khalf * 8;
            float4 h0 = *(const float4*)(hrow + kk);
            float4 h1 = *(const float4*)(hrow + kk + 4);
            float4 s0 = *(const float4*)(scale + kk);
            float4 s1 = *(const float4*)(scale + kk + 4);
            float4 t0 = *(const float4*)(shift + kk);
            float4 t1 = *(const float4*)(shift + kk + 4);
            __bf16* d = af.e + half * 8;
            d[0] = (__bf16)fmaf(h0.x, s0.x, t0.x);
            d[1] = (__bf16)fmaf(h0.y, s0.y, t0.y);
            d[2] = (__bf16)fmaf(h0.z, s0.z, t0.z);
            d[3] = (__bf16)fmaf(h0.w, s0.w, t0.w);
            d[4] = (__bf16)fmaf(h1.x, s1.x, t1.x);
            d[5] = (__bf16)fmaf(h1.y, s1.y, t1.y);
            d[6] = (__bf16)fmaf(h1.z, s1.z, t1.z);
            d[7] = (__bf16)fmaf(h1.w, s1.w, t1.w);
        }
#pragma unroll
        for (int j = 0; j < 2; ++j) {
            const __bf16* wrow = Wob + (size_t)(j * 16 + lane15) * NH + k0;
            acc[j] = wmma_bf16(af.v, frag_B(wrow, khalf), acc[j]);
        }
    }

#pragma unroll
    for (int j = 0; j < 2; ++j) {
        int n = j * 16 + lane15;
        if (n < NV) {
            float bias = bho[n];
            v8f c = acc[j];
#pragma unroll
            for (int r = 0; r < 8; ++r) {
                int m = mrow0 + khalf * 8 + r;
                out[(size_t)m * (NT * NV) + (size_t)t * NV + n] = c[r] + bias;
            }
        }
    }
}

// ------------------------------------------------------------------- launcher
extern "C" void kernel_launch(void* const* d_in, const int* in_sizes, int n_in,
                              void* d_out, int out_size, void* d_ws, size_t ws_size,
                              hipStream_t stream) {
    const int*   x    = (const int*)d_in[0];
    const float* emb  = (const float*)d_in[1];
    const float* W_hh = (const float*)d_in[2];
    const float* b_hh = (const float*)d_in[3];
    const float* W_ho = (const float*)d_in[4];
    const float* b_ho = (const float*)d_in[5];
    const float* gamma = (const float*)d_in[6];
    const float* beta  = (const float*)d_in[7];
    float* out = (float*)d_out;

    char* ws = (char*)d_ws;
    __bf16* Whh_b = (__bf16*)ws;                                  // 2 MB
    __bf16* Who_b = (__bf16*)(ws + 2u * 1024 * 1024);             // 64 KB
    float* hA = (float*)(ws + 2u * 1024 * 1024 + 64 * 1024);      // 8 MB
    float* hB = hA + (size_t)NB * NH;                             // 8 MB
    float* scale = hB + (size_t)NB * NH;
    float* shift = scale + NH;

    init_ws<<<8192, 256, 0, stream>>>(W_hh, W_ho, Whh_b, Who_b, hA);

    float* hin = hA;
    float* hout = hB;
    for (int t = 0; t < NT; ++t) {
        gemm_hh<<<dim3(NB / 128, NH / 128), 256, 0, stream>>>(hin, hout, x, t, emb,
                                                              Whh_b, b_hh);
        bn_stats<<<NH / 32, 256, 0, stream>>>(hout, gamma, beta, scale, shift);
        out_gemm<<<NB / 128, 256, 0, stream>>>(hout, scale, shift, Who_b, b_ho, out, t);
        float* tmp = hin; hin = hout; hout = tmp;
    }
}
